// InvariantPointAttention_50964081935576
// MI455X (gfx1250) — compile-verified, matrix operand order checked by reading, not run on hardware
//
#include <hip/hip_runtime.h>
#include <hip/hip_bf16.h>

typedef __attribute__((ext_vector_type(16))) __bf16 v16bf;
typedef __attribute__((ext_vector_type(8)))  float  v8f;

#define WMMA_BF16(a, b, c) \
  __builtin_amdgcn_wmma_f32_16x16x32_bf16(false, (a), false, (b), (short)0, (c), false, false)

// K-pair base offset inside a 32-wide K chunk for VGPR-pair index v (0..7)
// A (16x32, 16-bit): lane = half*16 + m ; VGPR v holds K = kb, kb+1 (lo, hi)
// B (32x16, 16-bit): lane = half*16 + n ; same K mapping (mirrored layout)
__device__ __forceinline__ int kpair_base(int v, int half) {
  return ((v >> 2) * 16) + (half * 8) + ((v & 3) * 2);
}

// ---------------------------------------------------------------------------
// K1: C[M x N] = A[M x K] @ B[K x N], fp32 in, bf16 WMMA, fp32 out.
// grid (M/16, N/16), block = 32 (one wave per tile).
// ---------------------------------------------------------------------------
__global__ __launch_bounds__(32) void gemm_proj(const float* __restrict__ A,
                                                const float* __restrict__ B,
                                                float* __restrict__ C,
                                                int N, int K) {
  const int i0 = blockIdx.x * 16, n0 = blockIdx.y * 16;
  const int lane = threadIdx.x & 31;
  const int half = lane >> 4, mn = lane & 15;
  v8f acc = {};
  for (int c0 = 0; c0 < K; c0 += 32) {
    v16bf a, b;
#pragma unroll
    for (int v = 0; v < 8; ++v) {
      const int kb = c0 + kpair_base(v, half);
      const float* ap = A + (size_t)(i0 + mn) * K + kb;
      a[2 * v]     = (__bf16)ap[0];
      a[2 * v + 1] = (__bf16)ap[1];
      const float* bp = B + (size_t)kb * N + (n0 + mn);
      b[2 * v]     = (__bf16)bp[0];
      b[2 * v + 1] = (__bf16)bp[N];
    }
    acc = WMMA_BF16(a, b, acc);
  }
#pragma unroll
  for (int r = 0; r < 8; ++r) {
    const int row = i0 + r + half * 8;
    C[(size_t)row * N + n0 + mn] = acc[r];
  }
}

// ---------------------------------------------------------------------------
// K2: per (h, n): bf16 per-head sq/sk/sv; global-frame points gq/gk/gv.
// ---------------------------------------------------------------------------
__global__ void frames_prep(const float* __restrict__ SQKV,  // [1024,576]
                            const float* __restrict__ VQK,   // [1024,288]
                            const float* __restrict__ VVp,   // [1024,288]
                            const float* __restrict__ rot,   // [1024,3,3]
                            const float* __restrict__ trans, // [1024,3]
                            __bf16* __restrict__ sq_b, __bf16* __restrict__ sk_b,
                            __bf16* __restrict__ sv_b,
                            float* __restrict__ gq, float* __restrict__ gk,
                            float* __restrict__ gv) {
  const int idx = blockIdx.x * blockDim.x + threadIdx.x;
  if (idx >= 12 * 1024) return;
  const int h = idx >> 10, n = idx & 1023;
  const float* srow = SQKV + (size_t)n * 576;
  const size_t hb = ((size_t)h * 1024 + n) * 16;
#pragma unroll
  for (int d = 0; d < 16; ++d) {
    sq_b[hb + d] = (__bf16)srow[h * 16 + d];
    sk_b[hb + d] = (__bf16)srow[192 + h * 16 + d];
    sv_b[hb + d] = (__bf16)srow[384 + h * 16 + d];
  }
  const float* R = rot + (size_t)n * 9;
  const float* t = trans + (size_t)n * 3;
  const float* vrow = VQK + (size_t)n * 288;
#pragma unroll
  for (int p = 0; p < 4; ++p) {
    float qx = vrow[h * 12 + p * 3 + 0], qy = vrow[h * 12 + p * 3 + 1], qz = vrow[h * 12 + p * 3 + 2];
    float kx = vrow[144 + h * 12 + p * 3 + 0], ky = vrow[144 + h * 12 + p * 3 + 1], kz = vrow[144 + h * 12 + p * 3 + 2];
    const size_t ob = ((size_t)h * 1024 + n) * 12 + p * 3;
#pragma unroll
    for (int c = 0; c < 3; ++c) {
      gq[ob + c] = R[c * 3 + 0] * qx + R[c * 3 + 1] * qy + R[c * 3 + 2] * qz + t[c];
      gk[ob + c] = R[c * 3 + 0] * kx + R[c * 3 + 1] * ky + R[c * 3 + 2] * kz + t[c];
    }
  }
  const float* wrow = VVp + (size_t)n * 288;
#pragma unroll
  for (int p = 0; p < 8; ++p) {
    float vx = wrow[h * 24 + p * 3 + 0], vy = wrow[h * 24 + p * 3 + 1], vz = wrow[h * 24 + p * 3 + 2];
    const size_t ob = ((size_t)h * 1024 + n) * 24 + p * 3;
#pragma unroll
    for (int c = 0; c < 3; ++c)
      gv[ob + c] = R[c * 3 + 0] * vx + R[c * 3 + 1] * vy + R[c * 3 + 2] * vz + t[c];
  }
}

// ---------------------------------------------------------------------------
// K3: logits[h, i0+ii, j0+jj] for all 12 heads. grid (64,64), block 128 (4 waves)
// ---------------------------------------------------------------------------
__global__ __launch_bounds__(128) void logits_kernel(
    const float* __restrict__ edge, const __bf16* __restrict__ sq_b,
    const __bf16* __restrict__ sk_b, const float* __restrict__ gq,
    const float* __restrict__ gk, const float* __restrict__ w_edge,
    const float* __restrict__ pw, float* __restrict__ logits) {
  __shared__ float tile[12][16][16];
  const int i0 = blockIdx.x * 16, j0 = blockIdx.y * 16;
  const int tid = threadIdx.x, lane = tid & 31, wave = tid >> 5;
  const int half = lane >> 4, mn = lane & 15;

  for (int e = tid; e < 12 * 256; e += 128) ((float*)tile)[e] = 0.f;
  __syncthreads();

  // ---- scalar q.k^T term (K=16 padded to 32) ----
  for (int h = wave; h < 12; h += 4) {
    v16bf a = {}, b = {};
#pragma unroll
    for (int v = 0; v < 8; ++v) {
      const int kb = kpair_base(v, half);
      if (kb < 16) {
        const __bf16* ap = sq_b + ((size_t)h * 1024 + (i0 + mn)) * 16 + kb;
        a[2 * v] = ap[0];  a[2 * v + 1] = ap[1];
        const __bf16* bp = sk_b + ((size_t)h * 1024 + (j0 + mn)) * 16 + kb;
        b[2 * v] = bp[0];  b[2 * v + 1] = bp[1];
      }
    }
    v8f c = {};
    c = WMMA_BF16(a, b, c);
#pragma unroll
    for (int r = 0; r < 8; ++r) tile[h][r + half * 8][mn] = 0.25f * c[r];  // D^-0.5
  }
  __syncthreads();

  // ---- edge bias: [16 j x 128] @ [128 x 12 heads] per ii (streams edge) ----
  // Hoist the loop-invariant w_edge B operands (4 K-chunks) into registers.
  v16bf bw[4];
#pragma unroll
  for (int cc = 0; cc < 4; ++cc) {
#pragma unroll
    for (int v = 0; v < 8; ++v) {
      const int kb = cc * 32 + kpair_base(v, half);
      bw[cc][2 * v]     = (mn < 12) ? (__bf16)w_edge[(size_t)kb * 12 + mn]       : (__bf16)0.0f;
      bw[cc][2 * v + 1] = (mn < 12) ? (__bf16)w_edge[(size_t)(kb + 1) * 12 + mn] : (__bf16)0.0f;
    }
  }
  for (int ii = wave * 4; ii < wave * 4 + 4; ++ii) {
    if (ii + 1 < 16)  // prefetch next row of the edge tile (global_prefetch_b8)
      __builtin_prefetch(edge + ((size_t)(i0 + ii + 1) * 1024 + (j0 + mn)) * 128, 0, 0);
    v8f c = {};
#pragma unroll
    for (int cc = 0; cc < 4; ++cc) {
      v16bf a;
#pragma unroll
      for (int v = 0; v < 8; ++v) {
        const int kb = cc * 32 + kpair_base(v, half);
        const float* ap = edge + ((size_t)(i0 + ii) * 1024 + (j0 + mn)) * 128 + kb;
        a[2 * v]     = (__bf16)ap[0];
        a[2 * v + 1] = (__bf16)ap[1];
      }
      c = WMMA_BF16(a, bw[cc], c);
    }
#pragma unroll
    for (int r = 0; r < 8; ++r)
      if (mn < 12) tile[mn][ii][r + half * 8] += c[r];  // (jj=r+8*half, head=mn)
  }
  __syncthreads();

  // ---- frame (point-distance) term + write logits ----
  const float wc = 0.11785113f;        // sqrt(2/PQ)/6
  const float rs3 = 0.57735027f;       // 3^-0.5
  for (int e = tid; e < 12 * 256; e += 128) {
    const int h = e >> 8, rem = e & 255, ii = rem >> 4, jj = rem & 15;
    const float* q = gq + ((size_t)h * 1024 + (i0 + ii)) * 12;
    const float* k = gk + ((size_t)h * 1024 + (j0 + jj)) * 12;
    float dist = 0.f;
#pragma unroll
    for (int p = 0; p < 4; ++p) {
      float dx = q[p * 3] - k[p * 3], dy = q[p * 3 + 1] - k[p * 3 + 1], dz = q[p * 3 + 2] - k[p * 3 + 2];
      dist += sqrtf(dx * dx + dy * dy + dz * dz);
    }
    const float val = tile[h][ii][jj] - wc * dist * pw[h];
    logits[((size_t)h * 1024 + (i0 + ii)) * 1024 + (j0 + jj)] = rs3 * val;
  }
}

// ---------------------------------------------------------------------------
// K4: row softmax over j (1024), emit bf16 attn. block per (h*1024+i).
// ---------------------------------------------------------------------------
__global__ __launch_bounds__(256) void softmax_kernel(const float* __restrict__ logits,
                                                      __bf16* __restrict__ attn_b) {
  const int row = blockIdx.x;
  const float* L = logits + (size_t)row * 1024;
  __shared__ float red[256];
  const int tid = threadIdx.x;
  float v[4];
  float m = -3.4e38f;
#pragma unroll
  for (int e = 0; e < 4; ++e) { v[e] = L[tid + e * 256]; m = fmaxf(m, v[e]); }
  red[tid] = m; __syncthreads();
  for (int s = 128; s > 0; s >>= 1) { if (tid < s) red[tid] = fmaxf(red[tid], red[tid + s]); __syncthreads(); }
  m = red[0]; __syncthreads();
  float sum = 0.f;
#pragma unroll
  for (int e = 0; e < 4; ++e) { v[e] = __expf(v[e] - m); sum += v[e]; }
  red[tid] = sum; __syncthreads();
  for (int s = 128; s > 0; s >>= 1) { if (tid < s) red[tid] += red[tid + s]; __syncthreads(); }
  const float inv = 1.f / red[0];
  __bf16* O = attn_b + (size_t)row * 1024;
#pragma unroll
  for (int e = 0; e < 4; ++e) O[tid + e * 256] = (__bf16)(v[e] * inv);
}

// ---------------------------------------------------------------------------
// K5: edge_out: per i, attn[12x1024] @ edge[i][1024x128]. block per i, 8 waves.
// The edge row (512 KB) is streamed through LDS with double-buffered
// global_load_async_to_lds_b128 (ASYNCcnt) so the DMA of chunk c+1 overlaps
// the WMMA consumption of chunk c. 2 x 16 KB staging buffers.
// ---------------------------------------------------------------------------
__global__ __launch_bounds__(256) void edge_out_kernel(const __bf16* __restrict__ attn_b,
                                                       const float* __restrict__ edge,
                                                       float* __restrict__ combined) {
  __shared__ float stage[2][32 * 128];
  const int i = blockIdx.x;
  const int tid = threadIdx.x;
  const int lane = tid & 31, wave = tid >> 5;
  const int half = lane >> 4, mn = lane & 15;
  const int d0 = wave * 16;

  const unsigned int lds0 = (unsigned int)(uintptr_t)(&stage[0][0]);
  const unsigned int lds1 = (unsigned int)(uintptr_t)(&stage[1][0]);
  const float* row_base = edge + (size_t)i * 1024 * 128;

  // Issue one 16 KB chunk (32 j-rows x 128 floats): 256 threads x 4 x 16B.
  auto issue = [&](int c, unsigned int ldsb) {
#pragma unroll
    for (int s = 0; s < 4; ++s) {
      const unsigned int off = (unsigned int)(tid + s * 256) * 16u;  // bytes
      const unsigned long long ga =
          (unsigned long long)(uintptr_t)(row_base + (size_t)c * 32 * 128) + off;
      asm volatile("global_load_async_to_lds_b128 %0, %1, off"
                   :: "v"(ldsb + off), "v"(ga)
                   : "memory");
    }
  };

  v8f acc = {};
  issue(0, lds0);
  for (int c = 0; c < 32; ++c) {
    asm volatile("s_wait_asynccnt 0" ::: "memory");
    __syncthreads();                       // chunk c resident in stage[c&1]
    if (c + 1 < 32) issue(c + 1, ((c + 1) & 1) ? lds1 : lds0);  // overlap DMA
    const float* sb = stage[c & 1];
    v16bf a, b;
#pragma unroll
    for (int v = 0; v < 8; ++v) {
      const int kl = kpair_base(v, half);   // 0..31 within chunk
      const int kb = c * 32 + kl;
      if (mn < 12) {
        const __bf16* ap = attn_b + ((size_t)mn * 1024 + i) * 1024 + kb;
        a[2 * v] = ap[0]; a[2 * v + 1] = ap[1];
      } else {
        a[2 * v] = (__bf16)0.0f; a[2 * v + 1] = (__bf16)0.0f;
      }
      b[2 * v]     = (__bf16)sb[kl * 128 + d0 + mn];
      b[2 * v + 1] = (__bf16)sb[(kl + 1) * 128 + d0 + mn];
    }
    acc = WMMA_BF16(a, b, acc);
    __syncthreads();                       // done reading stage[c&1]
  }
#pragma unroll
  for (int r = 0; r < 8; ++r) {
    const int h = r + half * 8;
    if (h < 12) combined[(size_t)i * 2112 + h * 128 + d0 + mn] = acc[r];
  }
}

// ---------------------------------------------------------------------------
// K6: attn @ sv (scalar_out) and attn @ gv (point agg). grid (64,12), 3 waves.
// ---------------------------------------------------------------------------
__global__ __launch_bounds__(96) void attnv_kernel(const __bf16* __restrict__ attn_b,
                                                   const __bf16* __restrict__ sv_b,
                                                   const float* __restrict__ gv,
                                                   float* __restrict__ combined,
                                                   float* __restrict__ agg) {
  const int i0 = blockIdx.x * 16, h = blockIdx.y;
  const int lane = threadIdx.x & 31, wave = threadIdx.x >> 5;
  const int half = lane >> 4, mn = lane & 15;
  v8f acc = {};
  for (int c0 = 0; c0 < 1024; c0 += 32) {
    v16bf a, b;
#pragma unroll
    for (int v = 0; v < 8; ++v) {
      const int kb = c0 + kpair_base(v, half);
      const __bf16* ap = attn_b + ((size_t)h * 1024 + (i0 + mn)) * 1024 + kb;
      a[2 * v] = ap[0]; a[2 * v + 1] = ap[1];
      if (wave == 0) {
        const __bf16* bp = sv_b + ((size_t)h * 1024 + kb) * 16 + mn;
        b[2 * v] = bp[0]; b[2 * v + 1] = bp[16];
      } else {
        const int n = (wave - 1) * 16 + mn;
        if (n < 24) {
          const float* bp = gv + ((size_t)h * 1024 + kb) * 24 + n;
          b[2 * v] = (__bf16)bp[0]; b[2 * v + 1] = (__bf16)bp[24];
        } else {
          b[2 * v] = (__bf16)0.0f; b[2 * v + 1] = (__bf16)0.0f;
        }
      }
    }
    acc = WMMA_BF16(a, b, acc);
  }
#pragma unroll
  for (int r = 0; r < 8; ++r) {
    const int row = i0 + r + half * 8;
    if (wave == 0) {
      combined[(size_t)row * 2112 + 1536 + h * 16 + mn] = acc[r];
    } else {
      const int n = (wave - 1) * 16 + mn;
      if (n < 24) agg[((size_t)h * 1024 + row) * 24 + n] = acc[r];
    }
  }
}

// ---------------------------------------------------------------------------
// K7: inverse frame R^T (agg - t) + norm -> combined vec_out block.
// ---------------------------------------------------------------------------
__global__ void vecout_kernel(const float* __restrict__ agg, const float* __restrict__ rot,
                              const float* __restrict__ trans, float* __restrict__ combined) {
  const int idx = blockIdx.x * blockDim.x + threadIdx.x;
  if (idx >= 12 * 1024 * 8) return;
  const int h = idx >> 13, rem = idx & 8191, n = rem >> 3, p = rem & 7;
  const float* a = agg + ((size_t)h * 1024 + n) * 24 + p * 3;
  const float* R = rot + (size_t)n * 9;
  const float* t = trans + (size_t)n * 3;
  const float x = a[0] - t[0], y = a[1] - t[1], z = a[2] - t[2];
  const float lx = R[0] * x + R[3] * y + R[6] * z;
  const float ly = R[1] * x + R[4] * y + R[7] * z;
  const float lz = R[2] * x + R[5] * y + R[8] * z;
  const float nr = sqrtf(lx * lx + ly * ly + lz * lz);
  float* o = combined + (size_t)n * 2112 + 1728 + h * 32 + p * 4;
  o[0] = nr; o[1] = lx; o[2] = ly; o[3] = lz;
}

// ---------------------------------------------------------------------------
// K8: out = node + combined[1024x2112] @ w_final[2112x384] + b_final
// ---------------------------------------------------------------------------
__global__ __launch_bounds__(32) void final_kernel(const float* __restrict__ combined,
                                                   const float* __restrict__ w_final,
                                                   const float* __restrict__ b_final,
                                                   const float* __restrict__ node,
                                                   float* __restrict__ out) {
  const int i0 = blockIdx.x * 16, n0 = blockIdx.y * 16;
  const int lane = threadIdx.x & 31;
  const int half = lane >> 4, mn = lane & 15;
  v8f acc = {};
  for (int c0 = 0; c0 < 2112; c0 += 32) {
    v16bf a, b;
#pragma unroll
    for (int v = 0; v < 8; ++v) {
      const int kb = c0 + kpair_base(v, half);
      const float* ap = combined + (size_t)(i0 + mn) * 2112 + kb;
      a[2 * v] = (__bf16)ap[0]; a[2 * v + 1] = (__bf16)ap[1];
      const float* bp = w_final + (size_t)kb * 384 + n0 + mn;
      b[2 * v] = (__bf16)bp[0]; b[2 * v + 1] = (__bf16)bp[384];
    }
    acc = WMMA_BF16(a, b, acc);
  }
#pragma unroll
  for (int r = 0; r < 8; ++r) {
    const int row = i0 + r + half * 8, col = n0 + mn;
    out[(size_t)row * 384 + col] = acc[r] + node[(size_t)row * 384 + col] + b_final[col];
  }
}

// ---------------------------------------------------------------------------
extern "C" void kernel_launch(void* const* d_in, const int* in_sizes, int n_in,
                              void* d_out, int out_size, void* d_ws, size_t ws_size,
                              hipStream_t stream) {
  const float* node    = (const float*)d_in[0];
  const float* edge    = (const float*)d_in[1];
  const float* rot     = (const float*)d_in[2];
  const float* trans   = (const float*)d_in[3];
  const float* w_sqkv  = (const float*)d_in[4];
  const float* w_vqk   = (const float*)d_in[5];
  const float* w_vv    = (const float*)d_in[6];
  const float* w_edge  = (const float*)d_in[7];
  const float* w_final = (const float*)d_in[8];
  const float* b_final = (const float*)d_in[9];
  const float* pw      = (const float*)d_in[10];
  float* out = (float*)d_out;

  char* W = (char*)d_ws;
  size_t cur = 0;
  auto alloc = [&](size_t bytes) {
    char* p = W + cur;
    cur += (bytes + 255) & ~(size_t)255;
    return p;
  };
  float*  SQKV   = (float*)alloc((size_t)1024 * 576 * 4);
  float*  VQK    = (float*)alloc((size_t)1024 * 288 * 4);
  float*  VVp    = (float*)alloc((size_t)1024 * 288 * 4);
  __bf16* sq_b   = (__bf16*)alloc((size_t)12 * 1024 * 16 * 2);
  __bf16* sk_b   = (__bf16*)alloc((size_t)12 * 1024 * 16 * 2);
  __bf16* sv_b   = (__bf16*)alloc((size_t)12 * 1024 * 16 * 2);
  float*  gq     = (float*)alloc((size_t)12 * 1024 * 12 * 4);
  float*  gk     = (float*)alloc((size_t)12 * 1024 * 12 * 4);
  float*  gv     = (float*)alloc((size_t)12 * 1024 * 24 * 4);
  float*  agg    = (float*)alloc((size_t)12 * 1024 * 24 * 4);
  float*  logits = (float*)alloc((size_t)12 * 1024 * 1024 * 4);
  __bf16* attn_b = (__bf16*)alloc((size_t)12 * 1024 * 1024 * 2);
  float*  comb   = (float*)alloc((size_t)1024 * 2112 * 4);

  gemm_proj<<<dim3(64, 36), 32, 0, stream>>>(node, w_sqkv, SQKV, 576, 384);
  gemm_proj<<<dim3(64, 18), 32, 0, stream>>>(node, w_vqk, VQK, 288, 384);
  gemm_proj<<<dim3(64, 18), 32, 0, stream>>>(node, w_vv, VVp, 288, 384);

  frames_prep<<<48, 256, 0, stream>>>(SQKV, VQK, VVp, rot, trans,
                                      sq_b, sk_b, sv_b, gq, gk, gv);

  logits_kernel<<<dim3(64, 64), 128, 0, stream>>>(edge, sq_b, sk_b, gq, gk,
                                                  w_edge, pw, logits);

  softmax_kernel<<<12 * 1024, 256, 0, stream>>>(logits, attn_b);

  edge_out_kernel<<<1024, 256, 0, stream>>>(attn_b, edge, comb);

  attnv_kernel<<<dim3(64, 12), 96, 0, stream>>>(attn_b, sv_b, gv, comb, agg);

  vecout_kernel<<<384, 256, 0, stream>>>(agg, rot, trans, comb);

  final_kernel<<<dim3(64, 24), 32, 0, stream>>>(comb, w_final, b_final, node, out);
}
